// Attention_63264868270688
// MI455X (gfx1250) — compile-verified, matrix operand order checked by reading, not run on hardware
//
#include <hip/hip_runtime.h>

typedef __attribute__((ext_vector_type(16))) _Float16     v16h;
typedef __attribute__((ext_vector_type(8)))  float        v8f;
typedef __attribute__((ext_vector_type(4)))  unsigned int v4u;
typedef __attribute__((ext_vector_type(8)))  int          v8i_;
typedef __attribute__((ext_vector_type(4)))  int          v4i_;

#define B_  2
#define T_  2048
#define C_  2048
#define H_  16
#define KV_ 4
#define D_  128
#define G_  4

// ---------------------------------------------------------------- convert
__global__ __launch_bounds__(256) void f32_to_f16_kernel(const float* __restrict__ in,
                                                         _Float16* __restrict__ out, int n) {
  int i = blockIdx.x * blockDim.x + threadIdx.x;
  int stride = gridDim.x * blockDim.x;
  for (; i < n; i += stride) out[i] = (_Float16)in[i];
}

// ---------------------------------------------------------------- TDM helper
// Issue a Tensor Data Mover load of a [64 rows x 32 halfs] f16 tile of a
// row-major [M,K] matrix (tile start = gsrc) into LDS at lds_off.
// D# packing per CDNA5 ISA 8.3/8.4: group0 = {count/lds/global/type},
// group1 = {data_size=2B, tensor_dim0=K, tensor_dim1=M, tile 32x64, stride0=K}.
// This toolchain carries the 6-arg builtin: (g0, g1, g2, g3, g4, cpol).
__device__ __forceinline__ void tdm_load_a_tile(const _Float16* gsrc, unsigned int lds_off,
                                                int M, int K) {
  unsigned long long ga = (unsigned long long)(size_t)gsrc;
  v4u g0 = { 1u,                                    // count=1 (valid user D#)
             lds_off,                               // lds_addr
             (unsigned int)ga,                      // global_addr[31:0]
             ((unsigned int)(ga >> 32) & 0x01ffffffu) | 0x80000000u }; // addr[56:32]|type=2
  v8i_ g1 = { (int)0x00010000,                                   // data_size=1 (2 bytes)
              (int)((K & 0xffff) << 16),                         // tensor_dim0[15:0]
              (int)(((K >> 16) & 0xffff) | ((M & 0xffff) << 16)),// dim0[31:16]|dim1[15:0]
              (int)(((M >> 16) & 0xffff) | (32 << 16)),          // dim1[31:16]|tile_dim0=32
              (int)64,                                           // tile_dim1=64 (tile_dim2=0)
              (int)K,                                            // tensor_dim0_stride[31:0]
              0, 0 };
  v4i_ gz4 = { 0, 0, 0, 0 };
  v8i_ gz8 = { 0, 0, 0, 0, 0, 0, 0, 0 };
  __builtin_amdgcn_tensor_load_to_lds(g0, g1, gz4, gz4, gz8, 0);
}

// ---------------------------------------------------------------- NT GEMM
// C[M,N] = A[M,K] (f16) * B[N,K]^T (f16), f32 out.
// Block tile 64(M) x 128(N); 8 waves, each computing a 32x32 sub-tile
// (4 WMMA accumulators, A/B fragment reuse). A tiles are DMA'd by the TDM
// into double-buffered LDS; B fragments stream from global (L2-resident).
__global__ __launch_bounds__(256) void gemm_nt_kernel(const _Float16* __restrict__ A,
                                                      const _Float16* __restrict__ Bm,
                                                      float* __restrict__ C,
                                                      int M, int N, int K) {
  __shared__ _Float16 As[2][64 * 32];
  const int lane  = threadIdx.x & 31;
  const int wave  = threadIdx.x >> 5;
  const int wy    = wave >> 2;          // 0..1 : 32-row sub-tile
  const int wx    = wave & 3;           // 0..3 : 32-col sub-tile
  const int lhalf = lane >> 4;
  const int lmod  = lane & 15;
  const int row0  = blockIdx.y * 64;
  const int col0  = blockIdx.x * 128 + wx * 32;

  const unsigned int lds0 = (unsigned int)(size_t)(&As[0][0]);
  const unsigned int lds1 = (unsigned int)(size_t)(&As[1][0]);

  v8f acc00 = {}, acc01 = {}, acc10 = {}, acc11 = {};

  const _Float16* Abase = A + (size_t)row0 * K;
  const int nk = K >> 5;

  if (wave == 0) tdm_load_a_tile(Abase, lds0, M, K);   // prefetch tile 0

  for (int ik = 0; ik < nk; ++ik) {
    const int kk = ik << 5;
    if (wave == 0) {
      if (ik + 1 < nk) {
        // double buffer: kick tile ik+1, wait for tile ik only
        tdm_load_a_tile(Abase + (ik + 1) * 32, (ik & 1) ? lds0 : lds1, M, K);
        __builtin_amdgcn_s_wait_tensorcnt((short)1);
      } else {
        __builtin_amdgcn_s_wait_tensorcnt((short)0);
      }
    }
    __syncthreads();

    const v16h* Asv = (const v16h*)As[ik & 1];
    v16h af0 = Asv[(32 * wy + lmod) * 2 + lhalf];
    v16h af1 = Asv[(32 * wy + 16 + lmod) * 2 + lhalf];

    const _Float16* b0p = Bm + (size_t)(col0 + lmod) * K + kk + lhalf * 16;
    const _Float16* b1p = Bm + (size_t)(col0 + 16 + lmod) * K + kk + lhalf * 16;
    v16h bf0 = *(const v16h*)b0p;
    v16h bf1 = *(const v16h*)b1p;
    __builtin_prefetch((const void*)(b0p + 32), 0, 0);
    __builtin_prefetch((const void*)(b1p + 32), 0, 0);

    acc00 = __builtin_amdgcn_wmma_f32_16x16x32_f16(false, af0, false, bf0, (short)0, acc00, false, false);
    acc01 = __builtin_amdgcn_wmma_f32_16x16x32_f16(false, af0, false, bf1, (short)0, acc01, false, false);
    acc10 = __builtin_amdgcn_wmma_f32_16x16x32_f16(false, af1, false, bf0, (short)0, acc10, false, false);
    acc11 = __builtin_amdgcn_wmma_f32_16x16x32_f16(false, af1, false, bf1, (short)0, acc11, false, false);
    __syncthreads();
  }

  // C/D layout: VGPR r, lane l -> (m = r + 8*(l/16), n = l%16)
#pragma unroll
  for (int r = 0; r < 8; ++r) {
    int m0 = row0 + 32 * wy + r + 8 * lhalf;
    int m1 = m0 + 16;
    C[(size_t)m0 * N + col0 + lmod]      = acc00[r];
    C[(size_t)m0 * N + col0 + 16 + lmod] = acc01[r];
    C[(size_t)m1 * N + col0 + lmod]      = acc10[r];
    C[(size_t)m1 * N + col0 + 16 + lmod] = acc11[r];
  }
}

// ---------------------------------------------------------------- RoPE (q)
// in : qf  [B,T,H,D] f32 (GEMM output),  out: qh [B,H,T,D] f16
__global__ __launch_bounds__(256) void rope_q_kernel(const float* __restrict__ qf,
                                                     _Float16* __restrict__ qh, int n) {
  int i0 = blockIdx.x * blockDim.x + threadIdx.x;
  int stride = gridDim.x * blockDim.x;
  for (int idx = i0; idx < n; idx += stride) {
    int d = idx & (D_ - 1);
    int h = (idx >> 7) & (H_ - 1);
    int t = (idx >> 11) & (T_ - 1);
    int b = idx >> 22;
    float x0 = qf[idx];
    float xr = (d < D_ / 2) ? -qf[idx + D_ / 2] : qf[idx - D_ / 2];
    int   di  = d & (D_ / 2 - 1);
    float inv = __powf(10000.0f, -(float)di * (2.0f / (float)D_));
    float s, c;
    __sincosf((float)t * inv, &s, &c);
    qh[(((size_t)(b * H_ + h)) * T_ + t) * D_ + d] = (_Float16)(x0 * c + xr * s);
  }
}

// ---------------------------------------------------------------- RoPE (k) + V transpose
// in : kvf [B,T,2*KV,D] f32;  out: kh [B,KV,T,D] f16 (RoPE'd), vT [B,KV,D,T] f16
__global__ __launch_bounds__(256) void rope_kv_kernel(const float* __restrict__ kvf,
                                                      _Float16* __restrict__ kh,
                                                      _Float16* __restrict__ vT, int n) {
  int i0 = blockIdx.x * blockDim.x + threadIdx.x;
  int stride = gridDim.x * blockDim.x;
  for (int idx = i0; idx < n; idx += stride) {
    int d = idx & (D_ - 1);
    int t = (idx >> 7) & (T_ - 1);
    int j = (idx >> 18) & (KV_ - 1);
    int b = idx >> 20;
    size_t src = (((size_t)(b * T_ + t)) * (2 * KV_) + j) * D_ + d;
    float kx = kvf[src];
    float kr = (d < D_ / 2) ? -kvf[src + D_ / 2] : kvf[src - D_ / 2];
    int   di  = d & (D_ / 2 - 1);
    float inv = __powf(10000.0f, -(float)di * (2.0f / (float)D_));
    float s, c;
    __sincosf((float)t * inv, &s, &c);
    kh[(((size_t)(b * KV_ + j)) * T_ + t) * D_ + d] = (_Float16)(kx * c + kr * s);
    float vx = kvf[src + (size_t)KV_ * D_];
    vT[(((size_t)(b * KV_ + j)) * D_ + d) * T_ + t] = (_Float16)vx;
  }
}

// ---------------------------------------------------------------- flash attention
// 8 independent waves per block, one 16-query tile each. Key tiles of 32 with
// online softmax; P staged through a per-wave LDS slab (DS is in-order per
// wave, so a wave-local s_wait_dscnt replaces any block barrier).
__global__ __launch_bounds__(256) void flash_attn_kernel(const _Float16* __restrict__ qh,
                                                         const _Float16* __restrict__ kh,
                                                         const _Float16* __restrict__ vT,
                                                         _Float16* __restrict__ yh) {
  __shared__ _Float16 Pt[8][16 * 32];
  const int wave  = threadIdx.x >> 5;
  const int lane  = threadIdx.x & 31;
  const int lhalf = lane >> 4;
  const int lmod  = lane & 15;
  const int qt  = blockIdx.x * 8 + wave;
  const int h   = blockIdx.y;
  const int b   = blockIdx.z;
  const int kvh = h / G_;
  _Float16* Ptw = Pt[wave];

  const _Float16* Qbase = qh + (((size_t)(b * H_ + h)) * T_ + qt * 16) * D_;
  v16h aq[4];
#pragma unroll
  for (int kc = 0; kc < 4; ++kc)
    aq[kc] = *(const v16h*)(Qbase + (size_t)lmod * D_ + kc * 32 + lhalf * 16);

  v8f zero = {};
  v8f oacc[8];
#pragma unroll
  for (int i = 0; i < 8; ++i) oacc[i] = zero;
  float mrow[8], lrow[8];
#pragma unroll
  for (int r = 0; r < 8; ++r) { mrow[r] = -__builtin_inff(); lrow[r] = 0.0f; }

  const _Float16* Kbase = kh + ((size_t)(b * KV_ + kvh)) * T_ * D_;
  const _Float16* Vbase = vT + ((size_t)(b * KV_ + kvh)) * D_ * T_;
  const float scale = 0.08838834764831845f;   // 1/sqrt(128)
  const int   nj    = ((qt * 16 + 15) >> 5) + 1;

  for (int jt = 0; jt < nj; ++jt) {
    const int key0 = jt * 32;

    // ---- S = Q K^T over the 16x32 key tile (8 WMMAs over D=128)
    v8f s0 = zero, s1 = zero;
#pragma unroll
    for (int kc = 0; kc < 4; ++kc) {
      v16h bk0 = *(const v16h*)(Kbase + (size_t)(key0 + lmod) * D_ + kc * 32 + lhalf * 16);
      v16h bk1 = *(const v16h*)(Kbase + (size_t)(key0 + 16 + lmod) * D_ + kc * 32 + lhalf * 16);
      s0 = __builtin_amdgcn_wmma_f32_16x16x32_f16(false, aq[kc], false, bk0, (short)0, s0, false, false);
      s1 = __builtin_amdgcn_wmma_f32_16x16x32_f16(false, aq[kc], false, bk1, (short)0, s1, false, false);
    }

    // ---- online softmax (row stats reduced across the 16-lane half)
    float p0[8], p1[8];
#pragma unroll
    for (int r = 0; r < 8; ++r) {
      const int m    = r + 8 * lhalf;
      const int qidx = qt * 16 + m;
      float a  = (key0 + lmod <= qidx)      ? s0[r] * scale : -__builtin_inff();
      float bb = (key0 + 16 + lmod <= qidx) ? s1[r] * scale : -__builtin_inff();
      float tmax = fmaxf(a, bb);
#pragma unroll
      for (int msk = 1; msk < 16; msk <<= 1) tmax = fmaxf(tmax, __shfl_xor(tmax, msk, 32));
      float newm = fmaxf(mrow[r], tmax);
      float corr = __expf(mrow[r] - newm);
      float e0 = __expf(a - newm);
      float e1 = __expf(bb - newm);
      float tsum = e0 + e1;
#pragma unroll
      for (int msk = 1; msk < 16; msk <<= 1) tsum += __shfl_xor(tsum, msk, 32);
      lrow[r] = lrow[r] * corr + tsum;
      mrow[r] = newm;
#pragma unroll
      for (int ds2 = 0; ds2 < 8; ++ds2) oacc[ds2][r] *= corr;
      p0[r] = e0;
      p1[r] = e1;
    }

    // ---- P: C-layout -> per-wave LDS slab -> A-layout fragment
    asm volatile("" ::: "memory");
#pragma unroll
    for (int r = 0; r < 8; ++r) {
      const int m = r + 8 * lhalf;
      Ptw[m * 32 + lmod]      = (_Float16)p0[r];
      Ptw[m * 32 + 16 + lmod] = (_Float16)p1[r];
    }
    asm volatile("s_wait_dscnt 0x0" ::: "memory");   // DS in-order per wave
    v16h ap = *(const v16h*)(Ptw + lmod * 32 + lhalf * 16);

    // ---- O += P V   (V^T stored as [D,T] -> contiguous B fragments)
#pragma unroll
    for (int ds2 = 0; ds2 < 8; ++ds2) {
      const int d = ds2 * 16 + lmod;
      v16h bv = *(const v16h*)(Vbase + (size_t)d * T_ + key0 + lhalf * 16);
      oacc[ds2] = __builtin_amdgcn_wmma_f32_16x16x32_f16(false, ap, false, bv, (short)0, oacc[ds2], false, false);
    }
  }

  // ---- normalize and store y as [B, T, H*D] (= input layout for O-proj)
#pragma unroll
  for (int r = 0; r < 8; ++r) {
    const int   m    = r + 8 * lhalf;
    const float invl = 1.0f / lrow[r];
    const size_t trow = ((size_t)b * T_ + qt * 16 + m) * C_ + (size_t)h * D_;
#pragma unroll
    for (int ds2 = 0; ds2 < 8; ++ds2)
      yh[trow + ds2 * 16 + lmod] = (_Float16)(oacc[ds2][r] * invl);
  }
}

// ---------------------------------------------------------------- launcher
extern "C" void kernel_launch(void* const* d_in, const int* in_sizes, int n_in,
                              void* d_out, int out_size, void* d_ws, size_t ws_size,
                              hipStream_t stream) {
  const float* x   = (const float*)d_in[0];
  const float* Wq  = (const float*)d_in[1];
  const float* Wkv = (const float*)d_in[2];
  const float* Wo  = (const float*)d_in[3];
  float* out = (float*)d_out;

  char* wp = (char*)d_ws;
  auto walloc = [&](size_t bytes) -> char* {
    char* r = wp;
    wp += (bytes + 255) & ~(size_t)255;
    return r;
  };

  const size_t nX   = (size_t)B_ * T_ * C_;            // 8,388,608
  const size_t nWq  = (size_t)C_ * C_;                 // 4,194,304
  const size_t nWkv = (size_t)2 * KV_ * D_ * C_;       // 2,097,152
  const size_t nKV  = (size_t)B_ * KV_ * T_ * D_;      // 2,097,152

  _Float16* xh   = (_Float16*)walloc(nX * 2);
  _Float16* Wqh  = (_Float16*)walloc(nWq * 2);
  _Float16* Wkvh = (_Float16*)walloc(nWkv * 2);
  _Float16* Woh  = (_Float16*)walloc(nWq * 2);
  float*    qf   = (float*)walloc(nX * 4);             // [B,T,H,D] f32
  float*    kvf  = (float*)walloc((size_t)B_ * T_ * 2 * KV_ * D_ * 4);
  _Float16* qhh  = (_Float16*)walloc(nX * 2);          // [B,H,T,D]
  _Float16* khh  = (_Float16*)walloc(nKV * 2);         // [B,KV,T,D]
  _Float16* vT   = (_Float16*)walloc(nKV * 2);         // [B,KV,D,T]
  _Float16* yh   = (_Float16*)walloc(nX * 2);          // [B,T,C]

  // 1) downconvert inputs to f16
  f32_to_f16_kernel<<<2048, 256, 0, stream>>>(x, xh, (int)nX);
  f32_to_f16_kernel<<<2048, 256, 0, stream>>>(Wq, Wqh, (int)nWq);
  f32_to_f16_kernel<<<1024, 256, 0, stream>>>(Wkv, Wkvh, (int)nWkv);
  f32_to_f16_kernel<<<2048, 256, 0, stream>>>(Wo, Woh, (int)nWq);

  // 2) QKV projections (NT WMMA GEMMs, TDM-staged A tiles)
  dim3 gq(C_ / 128, (B_ * T_) / 64);
  gemm_nt_kernel<<<gq, 256, 0, stream>>>(xh, Wqh, qf, B_ * T_, C_, C_);
  dim3 gkv((2 * KV_ * D_) / 128, (B_ * T_) / 64);
  gemm_nt_kernel<<<gkv, 256, 0, stream>>>(xh, Wkvh, kvf, B_ * T_, 2 * KV_ * D_, C_);

  // 3) RoPE + layout transforms
  rope_q_kernel<<<4096, 256, 0, stream>>>(qf, qhh, (int)nX);
  rope_kv_kernel<<<2048, 256, 0, stream>>>(kvf, khh, vT, (int)nKV);

  // 4) causal flash attention (8 query-tile waves per block)
  dim3 ga(T_ / 16 / 8, H_, B_);
  flash_attn_kernel<<<ga, 256, 0, stream>>>(qhh, khh, vT, yh);

  // 5) output projection -> f32 result
  gemm_nt_kernel<<<gq, 256, 0, stream>>>(yh, Woh, out, B_ * T_, C_, C_);
}
